// MultiHeadAttention_25881472926641
// MI455X (gfx1250) — matrix operand order repaired and checked by reading, not compiled, not run on hardware
//
#include <hip/hip_runtime.h>
#include <hip/hip_bf16.h>

// ---------------------------------------------------------------- constants
static constexpr int Bn = 2;        // batch
static constexpr int T  = 2048;     // sequence
static constexpr int E  = 1024;     // embed
static constexpr int H  = 16;       // heads
static constexpr int D  = 64;       // head dim
static constexpr float SCALE = 0.125f;  // D^-0.5

typedef __attribute__((ext_vector_type(16))) __bf16 bf16x16;
typedef __attribute__((ext_vector_type(8)))  __bf16 bf16x8;
typedef __attribute__((ext_vector_type(8)))  float  f32x8;

// ---------------------------------------------------------------- helpers
__device__ __forceinline__ bf16x16 pack16(bf16x8 lo, bf16x8 hi) {
    bf16x16 r;
#pragma unroll
    for (int i = 0; i < 8; ++i) { r[i] = lo[i]; r[i + 8] = hi[i]; }
    return r;
}

// A-matrix 16x32 (M x K) from row-major source (ld = row stride in elems).
// ISA layout: lanes 0-15 hold K 0-7 / 16-23, lanes 16-31 hold K 8-15 / 24-31.
__device__ __forceinline__ bf16x16 load_a(const __bf16* base, int ld,
                                          int row0, int k0, int lane) {
    const int m  = lane & 15;
    const int ko = (lane >> 4) << 3;
    const __bf16* p = base + (size_t)(row0 + m) * ld + k0 + ko;
    return pack16(*(const bf16x8*)p, *(const bf16x8*)(p + 16));
}

// B-matrix 32x16 (K x N) from source stored K-contiguous per column
// (ld = column stride in elems). lanes 0-15: K 0-7 / 16-23 of col n; lanes 16-31: K 8-15 / 24-31.
__device__ __forceinline__ bf16x16 load_b(const __bf16* base, int ld,
                                          int col0, int k0, int lane) {
    const int n  = lane & 15;
    const int kg = (lane >> 4) << 3;
    const __bf16* p = base + (size_t)(col0 + n) * ld + k0 + kg;
    return pack16(*(const bf16x8*)p, *(const bf16x8*)(p + 16));
}

__device__ __forceinline__ f32x8 wmma_bf16(bf16x16 a, bf16x16 b, f32x8 c) {
    return __builtin_amdgcn_wmma_f32_16x16x32_bf16(false, a, false, b,
                                                   (short)0, c, false, false);
}

// ---------------------------------------------------------------- converts
__global__ void cvt_x_kernel(const float* __restrict__ x,
                             __bf16* __restrict__ xb, int n) {
    int i = blockIdx.x * blockDim.x + threadIdx.x;
    if (i < n) xb[i] = (__bf16)x[i];
}

// Wt[(proj,h,d,e)] = w_proj[h][e][d]   (weights transposed, bf16)
__global__ void cvt_w_kernel(const float* __restrict__ qw,
                             const float* __restrict__ kw,
                             const float* __restrict__ vw,
                             __bf16* __restrict__ Wt) {
    int i = blockIdx.x * blockDim.x + threadIdx.x;
    if (i >= 3 * H * D * E) return;
    int e = i % E;  int r = i / E;
    int d = r % D;  r /= D;
    int h = r % H;  r /= H;                 // r = proj
    const float* src = (r == 0) ? qw : (r == 1) ? kw : vw;
    Wt[i] = (__bf16)src[((size_t)h * E + e) * D + d];
}

// ---------------------------------------------------------------- projection
// grid = (B*T/64, H, 3); block = 128 (4 waves); wave computes 16x64 of Q/K/V.
// Ping-pong double buffering (2x unrolled k-loop): no rotation copies.
__global__ __launch_bounds__(128) void proj_kernel(
        const __bf16* __restrict__ Xbf, const __bf16* __restrict__ Wt,
        const float* __restrict__ qb, const float* __restrict__ kb,
        const float* __restrict__ vb,
        __bf16* __restrict__ Q, __bf16* __restrict__ K,
        __bf16* __restrict__ Vt) {
    const int lane = threadIdx.x & 31;
    const int wave = threadIdx.x >> 5;
    const int h    = blockIdx.y;
    const int proj = blockIdx.z;
    const int row0 = blockIdx.x * 64 + wave * 16;     // global row in [0, B*T)
    const __bf16* wbase = Wt + ((size_t)(proj * H + h)) * D * E;
    const float*  bias  = ((proj == 0) ? qb : (proj == 1) ? kb : vb) + h * D;

    f32x8 acc[4] = {};
    bf16x16 a0 = load_a(Xbf, E, row0, 0, lane);
    bf16x16 b0[4];
#pragma unroll
    for (int nt = 0; nt < 4; ++nt) b0[nt] = load_b(wbase, E, nt * 16, 0, lane);

    for (int k0 = 0; k0 < E; k0 += 64) {
        // prefetch odd step (k0+32) into buffer 1
        bf16x16 a1 = load_a(Xbf, E, row0, k0 + 32, lane);
        bf16x16 b1[4];
#pragma unroll
        for (int nt = 0; nt < 4; ++nt) b1[nt] = load_b(wbase, E, nt * 16, k0 + 32, lane);
        // consume even step from buffer 0
#pragma unroll
        for (int nt = 0; nt < 4; ++nt) acc[nt] = wmma_bf16(a0, b0[nt], acc[nt]);
        // prefetch next even step (k0+64) straight into buffer 0
        if (k0 + 64 < E) {
            a0 = load_a(Xbf, E, row0, k0 + 64, lane);
#pragma unroll
            for (int nt = 0; nt < 4; ++nt) b0[nt] = load_b(wbase, E, nt * 16, k0 + 64, lane);
        }
        // consume odd step from buffer 1
#pragma unroll
        for (int nt = 0; nt < 4; ++nt) acc[nt] = wmma_bf16(a1, b1[nt], acc[nt]);
    }

    const int n  = lane & 15;
    const int mb = (lane >> 4) << 3;
    const int bi = row0 / T;
    const int t0 = row0 % T;
    const size_t bh = (size_t)bi * H + h;
#pragma unroll
    for (int nt = 0; nt < 4; ++nt) {
        const int d  = nt * 16 + n;
        const float bv = bias[d];
#pragma unroll
        for (int i = 0; i < 8; ++i) {
            const int t = t0 + mb + i;
            __bf16 val = (__bf16)(acc[nt][i] + bv);
            if (proj == 0)      Q [(bh * T + t) * D + d] = val;
            else if (proj == 1) K [(bh * T + t) * D + d] = val;
            else                Vt[(bh * D + d) * T + t] = val;   // V transposed
        }
    }
}

// ---------------------------------------------------------------- colsum pass
// l[bh][s] = sum_{t>=s} exp(q_t . k_s * scale)   (query-axis softmax denom)
// grid = (T/64, B*H); block = 128; wave owns 16 columns. K tiles hoisted;
// two t-tiles per iteration with fixed ping-pong buffers (no copies).
__global__ __launch_bounds__(128) void colsum_kernel(
        const __bf16* __restrict__ Q, const __bf16* __restrict__ K,
        float* __restrict__ l) {
    const int lane = threadIdx.x & 31;
    const int wave = threadIdx.x >> 5;
    const int bh   = blockIdx.y;
    const int s0   = blockIdx.x * 64 + wave * 16;
    const __bf16* Qb = Q + (size_t)bh * T * D;
    const __bf16* Kb = K + (size_t)bh * T * D;
    const int n  = lane & 15;
    const int mb = (lane >> 4) << 3;
    const int s  = s0 + n;

    const bf16x16 bk0 = load_b(Kb, D, s0, 0, lane);   // loop-invariant K^T tiles
    const bf16x16 bk1 = load_b(Kb, D, s0, 32, lane);
    bf16x16 aA0 = load_a(Qb, D, s0, 0, lane);
    bf16x16 aA1 = load_a(Qb, D, s0, 32, lane);
    bf16x16 aB0, aB1;

    float part = 0.f;
    for (int t0 = s0; t0 < T; t0 += 32) {             // causal: only t >= s
        const bool hasB = (t0 + 16) < T;
        if (hasB) {
            aB0 = load_a(Qb, D, t0 + 16, 0, lane);
            aB1 = load_a(Qb, D, t0 + 16, 32, lane);
        }
        {
            f32x8 c = {};
            c = wmma_bf16(aA0, bk0, c);
            c = wmma_bf16(aA1, bk1, c);
#pragma unroll
            for (int i = 0; i < 8; ++i) {
                const int t = t0 + mb + i;
                part += (t >= s) ? __expf(c[i] * SCALE) : 0.f;
            }
        }
        if (t0 + 32 < T) {
            aA0 = load_a(Qb, D, t0 + 32, 0, lane);
            aA1 = load_a(Qb, D, t0 + 32, 32, lane);
        }
        if (hasB) {
            f32x8 c = {};
            c = wmma_bf16(aB0, bk0, c);
            c = wmma_bf16(aB1, bk1, c);
#pragma unroll
            for (int i = 0; i < 8; ++i) {
                const int t = t0 + 16 + mb + i;
                part += (t >= s) ? __expf(c[i] * SCALE) : 0.f;
            }
        }
    }
    part += __shfl_xor(part, 16, 32);          // combine m-halves (lane ^ 16)
    if (lane < 16) l[(size_t)bh * T + s0 + lane] = part;
}

// ---------------------------------------------------------------- output pass
// out[t,:] = sum_s exp(S[t,s])/l[s] * V[s,:]; grid = (T/64, B*H); block = 128.
// Q tiles loop-invariant; trip count (tb+64)/32 is even -> exact 2x unroll
// with bmA/bmB ping-pong (no rotation copies).
__global__ __launch_bounds__(128) void attn_kernel(
        const __bf16* __restrict__ Q, const __bf16* __restrict__ K,
        const __bf16* __restrict__ Vt, const float* __restrict__ l,
        float* __restrict__ out) {
    __shared__ __align__(16) __bf16 plds[4][16][40];   // per-wave P tile, stride 40
    const int lane = threadIdx.x & 31;
    const int wave = threadIdx.x >> 5;
    const int bh   = blockIdx.y;
    const int b    = bh / H, h = bh % H;
    const int tb   = blockIdx.x * 64;
    const int r0   = tb + wave * 16;
    const __bf16* Qb = Q  + (size_t)bh * T * D;
    const __bf16* Kb = K  + (size_t)bh * T * D;
    const __bf16* Vb = Vt + (size_t)bh * D * T;
    const float*  lb = l  + (size_t)bh * T;
    const int n  = lane & 15;
    const int mb = (lane >> 4) << 3;
    const int ko = (lane >> 4) << 3;

    const bf16x16 aq0 = load_a(Qb, D, r0, 0, lane);    // loop-invariant Q tiles
    const bf16x16 aq1 = load_a(Qb, D, r0, 32, lane);

    f32x8 acc[4] = {};
    const int send = tb + 64;                   // causal bound for this block

    // one 32-key step: uses bmc (K tiles for sb), optionally prefetches K tiles
    // for sb+32 into bmn; V tiles loaded up front so PV finds them resident.
    auto step = [&](int sb, bf16x16 (&bmc)[2][2], bf16x16 (&bmn)[2][2], bool pf) {
        bf16x16 bv[4];
#pragma unroll
        for (int nt = 0; nt < 4; ++nt) bv[nt] = load_b(Vb, T, nt * 16, sb, lane);
        if (pf) {
#pragma unroll
            for (int sub = 0; sub < 2; ++sub) {
                bmn[sub][0] = load_b(Kb, D, sb + 32 + sub * 16, 0, lane);
                bmn[sub][1] = load_b(Kb, D, sb + 32 + sub * 16, 32, lane);
            }
        }
        // S = Q K^T tile, exp()/l, write P stripe to wave-private LDS
#pragma unroll
        for (int sub = 0; sub < 2; ++sub) {
            f32x8 c = {};
            c = wmma_bf16(aq0, bmc[sub][0], c);
            c = wmma_bf16(aq1, bmc[sub][1], c);
            const int s = sb + sub * 16 + n;
            const float linv = __builtin_amdgcn_rcpf(lb[s]);
#pragma unroll
            for (int i = 0; i < 8; ++i) {
                const int t = r0 + mb + i;
                float p = (t >= s) ? __expf(c[i] * SCALE) * linv : 0.f;
                plds[wave][mb + i][sub * 16 + n] = (__bf16)p;  // wave-private
            }
        }
        // reload P as A-matrix (16x32) and multiply by V
        const __bf16* p = &plds[wave][lane & 15][ko];
        bf16x16 ap = pack16(*(const bf16x8*)p, *(const bf16x8*)(p + 16));
#pragma unroll
        for (int nt = 0; nt < 4; ++nt) acc[nt] = wmma_bf16(ap, bv[nt], acc[nt]);
    };

    bf16x16 bmA[2][2], bmB[2][2];
#pragma unroll
    for (int sub = 0; sub < 2; ++sub) {
        bmA[sub][0] = load_b(Kb, D, sub * 16, 0, lane);
        bmA[sub][1] = load_b(Kb, D, sub * 16, 32, lane);
    }
    for (int sb = 0; sb < send; sb += 64) {
        step(sb,      bmA, bmB, true);               // sb+32 < send always holds
        step(sb + 32, bmB, bmA, (sb + 64) < send);
    }

    // ---- store (B, T, H*D), head-major concat
    float* ob = out + (size_t)b * T * (H * D) + h * D;
#pragma unroll
    for (int nt = 0; nt < 4; ++nt) {
        const int d = nt * 16 + n;
#pragma unroll
        for (int i = 0; i < 8; ++i) {
            const int t = r0 + mb + i;
            ob[(size_t)t * (H * D) + d] = acc[nt][i];
        }
    }
}

// ---------------------------------------------------------------- launcher
extern "C" void kernel_launch(void* const* d_in, const int* in_sizes, int n_in,
                              void* d_out, int out_size, void* d_ws, size_t ws_size,
                              hipStream_t stream) {
    // input order: inputs, k_w, k_b, q_w, q_b, v_w, v_b
    const float* x  = (const float*)d_in[0];
    const float* kw = (const float*)d_in[1];
    const float* kb = (const float*)d_in[2];
    const float* qw = (const float*)d_in[3];
    const float* qb = (const float*)d_in[4];
    const float* vw = (const float*)d_in[5];
    const float* vb = (const float*)d_in[6];
    float* out = (float*)d_out;

    // workspace layout (bytes)
    uint8_t* ws = (uint8_t*)d_ws;
    __bf16* Xbf = (__bf16*)(ws);                                   // B*T*E bf16
    __bf16* Wt  = (__bf16*)(ws + 8388608);                         // 3*H*D*E bf16
    __bf16* Qm  = (__bf16*)(ws + 8388608 + 6291456);               // B*H*T*D bf16
    __bf16* Km  = (__bf16*)(ws + 8388608 + 6291456 + 8388608);
    __bf16* Vtm = (__bf16*)(ws + 8388608 + 6291456 + 2 * 8388608);
    float*  lm  = (float*)(ws + 8388608 + 6291456 + 3 * 8388608);  // B*H*T f32

    const int nx = Bn * T * E;                  // 4,194,304
    const int nw = 3 * H * D * E;               // 3,145,728
    cvt_x_kernel<<<(nx + 255) / 256, 256, 0, stream>>>(x, Xbf, nx);
    cvt_w_kernel<<<(nw + 255) / 256, 256, 0, stream>>>(qw, kw, vw, Wt);

    proj_kernel<<<dim3(Bn * T / 64, H, 3), 128, 0, stream>>>(
        Xbf, Wt, qb, kb, vb, Qm, Km, Vtm);

    colsum_kernel<<<dim3(T / 64, Bn * H), 128, 0, stream>>>(Qm, Km, lm);

    attn_kernel<<<dim3(T / 64, Bn * H), 128, 0, stream>>>(Qm, Km, Vtm, lm, out);
}
